// MACEGNN_28647431864803
// MI455X (gfx1250) — compile-verified, matrix operand order checked by reading, not run on hardware
//
#include <hip/hip_runtime.h>
#include <hip/hip_bf16.h>
#include <math.h>

// ---------------- problem constants ----------------
#define B_SZ     32
#define N_NODES  128
#define FCH      64          // hidden channels F
#define N_RBF    8
#define THREE_F  192         // 3*F
#define R_MAXC   5.0f
#define AVG_NEIGH 127.0f
#define LOG2E    1.4426950408889634f

typedef __attribute__((ext_vector_type(16))) _Float16 v16h;
typedef __attribute__((ext_vector_type(8)))  float    v8f;

// ---- shared-memory layout (bytes).  The weight staging region (prologue
// only) is aliased with the double-buffered sender-tile buffers. ----
#define SM_S0     0          // sS buf0: 4KB    (alias: sW1 @0 1KB, sW2 @1024 24KB)
#define SM_V0     4096       // sV buf0: 12KB
#define SM_S1     16384      // sS buf1: 4KB
#define SM_V1     20480      // sV buf1: 12KB
#define SM_B1     32768      // bias: 256B      (persistent)
#define SM_POS    33024      // positions: 1.5KB
#define SM_UNIT   34560      // unit vecs: 8 waves * 256B = 2KB
#define SM_H      36608      // h tiles: 8 waves * 2KB = 16KB
#define SM_TOTAL  52992

// K-slot mapping for 16-bit A/B fragments of v_wmma_*_16x16x32_*:
// lanes 0-15: VGPR0..3 hold K=0..7 (2/VGPR), VGPR4..7 hold K=16..23
// lanes16-31: same rows, K=8..15 and K=24..31
__device__ __forceinline__ int kslot(int p, int t, int h) {
    return (p < 4 ? 2 * p + t : 16 + 2 * (p - 4) + t) + 8 * h;
}

__device__ __forceinline__ float fast_rcp(float x)  { return __builtin_amdgcn_rcpf(x); }
__device__ __forceinline__ float fast_rsq(float x)  { return __builtin_amdgcn_rsqf(x); }
__device__ __forceinline__ float fast_exp2(float x) { return __builtin_amdgcn_exp2f(x); }

// Issue one async memory->LDS b128 transfer (ASYNCcnt tracked).
__device__ __forceinline__ void async_b128(unsigned lds_addr, const void* gaddr) {
    unsigned long long ga = (unsigned long long)gaddr;
    asm volatile("global_load_async_to_lds_b128 %0, %1, off"
                 :: "v"(lds_addr), "v"(ga) : "memory");
}

// ---------------------------------------------------------------------------
// Kernel 1: node init   s = embed[species] + glob @ glob_w ;  v = 0
// ---------------------------------------------------------------------------
__global__ void macegnn_init(const int* __restrict__ species,
                             const float* __restrict__ glob,
                             const float* __restrict__ embed,
                             const float* __restrict__ glob_w,
                             float* __restrict__ s_out,
                             float* __restrict__ v_out) {
    int idx = blockIdx.x * blockDim.x + threadIdx.x;
    if (idx >= B_SZ * N_NODES * FCH) return;
    int f = idx % FCH;
    int n = (idx / FCH) % N_NODES;
    int b = idx / (FCH * N_NODES);
    int sp = species[b * N_NODES + n];
    float acc = embed[sp * FCH + f];
    #pragma unroll
    for (int g = 0; g < 16; ++g)
        acc += glob[b * 16 + g] * glob_w[g * FCH + f];
    s_out[idx] = acc;
    v_out[idx * 3 + 0] = 0.0f;
    v_out[idx * 3 + 1] = 0.0f;
    v_out[idx * 3 + 2] = 0.0f;
}

// ---------------------------------------------------------------------------
// Kernel 2: interaction layer (WMMA hot loop, async double-buffered staging).
// grid = (32 batches, 16 receiver-chunks), block = 256 threads (8 waves),
// one receiver per wave.  Per sender-tile of 16: the block async-prefetches
// the NEXT tile's s (4KB) + v (12KB) into the alternate LDS buffer while
// computing the current one (4 WMMA phi@W1 + 24 WMMA h@W2 per wave, message
// accumulation fused in the D-register layout).  One barrier per tile.
// ---------------------------------------------------------------------------
__global__ __launch_bounds__(256) void macegnn_interact(
        const float* __restrict__ pos,      // [B,128,3]
        const float* __restrict__ s_in,     // [B,128,64]
        const float* __restrict__ v_in,     // [B,128,64,3]
        const float* __restrict__ w1,       // [8,64]   layer slice
        const float* __restrict__ b1,       // [64]
        const float* __restrict__ w2,       // [64,192] layer slice
        float* __restrict__ agg_s,
        float* __restrict__ agg_v) {
    __shared__ __align__(16) char smem[SM_TOTAL];

    _Float16* sW1  = (_Float16*)(smem);              // prologue only (aliased)
    _Float16* sW2  = (_Float16*)(smem + 1024);       // prologue only (aliased)
    float*    sB1  = (float*)(smem + SM_B1);
    float*    sPos = (float*)(smem + SM_POS);

    const int b      = blockIdx.x;
    const int rchunk = blockIdx.y;                   // 0..15
    const int tid    = threadIdx.x;
    const int wave   = tid >> 5;
    const int lane   = tid & 31;
    const int half   = lane >> 4;
    const int lmod   = lane & 15;

    float*    sUnitW = (float*)(smem + SM_UNIT) + wave * 64;      // 16*4 f32
    _Float16* sHW    = (_Float16*)(smem + SM_H) + wave * 16 * FCH;

    // ---- stage weights (f32 -> f16), bias and positions into LDS ----
    for (int i = tid; i < N_RBF * FCH; i += 256) sW1[i] = (_Float16)w1[i];
    for (int i = tid; i < FCH; i += 256)         sB1[i] = b1[i];
    for (int i = tid; i < FCH * THREE_F; i += 256) sW2[i] = (_Float16)w2[i];
    for (int i = tid; i < N_NODES * 3; i += 256) sPos[i] = pos[(size_t)b * N_NODES * 3 + i];
    __syncthreads();

    // ---- preload all B fragments into VGPRs (reused for every tile) ----
    v16h fw1[4];        // W1: 4 N-tiles, K = 8 (zero padded to 32)
    v16h fw2[12][2];    // W2: 12 N-tiles x 2 K-halves
    #pragma unroll
    for (int nt = 0; nt < 4; ++nt) {
        #pragma unroll
        for (int p = 0; p < 8; ++p) {
            #pragma unroll
            for (int t = 0; t < 2; ++t) {
                int k = kslot(p, t, half);
                fw1[nt][2 * p + t] =
                    (k < N_RBF) ? sW1[k * FCH + nt * 16 + lmod] : (_Float16)0.0f;
            }
        }
    }
    #pragma unroll
    for (int nt = 0; nt < 12; ++nt) {
        #pragma unroll
        for (int kh = 0; kh < 2; ++kh) {
            #pragma unroll
            for (int p = 0; p < 8; ++p) {
                #pragma unroll
                for (int t = 0; t < 2; ++t) {
                    int k = kslot(p, t, half) + kh * 32;
                    fw2[nt][kh][2 * p + t] = sW2[k * THREE_F + nt * 16 + lmod];
                }
            }
        }
    }
    // All waves done reading the weight region before it is overwritten by
    // the first async tile stage.
    asm volatile("s_wait_dscnt 0x0" ::: "memory");
    __syncthreads();

    // Gaussian RBF: exp(-(d-c_k)^2 / (2w^2))  ->  exp2(-(d-c_k)^2 * g2)
    const float g2    = LOG2E / (2.0f * (R_MAXC / N_RBF) * (R_MAXC / N_RBF));
    const float cstep = R_MAXC / (float)(N_RBF - 1);    // linspace(0, R, 8)
    const float SC    = 1.0f / AVG_NEIGH;
    const float* sbase = s_in + (size_t)b * N_NODES * FCH;
    const float* vbase = v_in + (size_t)b * N_NODES * FCH * 3;

    const unsigned ldsS[2] = { (unsigned)(size_t)(smem + SM_S0),
                               (unsigned)(size_t)(smem + SM_S1) };
    const unsigned ldsV[2] = { (unsigned)(size_t)(smem + SM_V0),
                               (unsigned)(size_t)(smem + SM_V1) };

    // ---- async prefetch of tile 0 into buffer 0 ----
    // sS: 256 x b128 (1 per thread); sV: 768 x b128 (3 per thread)
    {
        async_b128(ldsS[0] + tid * 16, (const char*)sbase + tid * 16);
        #pragma unroll
        for (int q = 0; q < 3; ++q)
            async_b128(ldsV[0] + (q * 256 + tid) * 16,
                       (const char*)vbase + (q * 256 + tid) * 16);
    }

    const int j = rchunk * 8 + wave;                    // this wave's receiver
    const float rx = sPos[j * 3 + 0];
    const float ry = sPos[j * 3 + 1];
    const float rz = sPos[j * 3 + 2];

    float acc_s[4]    = {0.f, 0.f, 0.f, 0.f};
    float acc_v[3][4] = {};

    #pragma unroll 1
    for (int tile = 0; tile < 8; ++tile) {
        const int cur = tile & 1;
        // staged data for this tile complete (own async) + visible (barrier)
        asm volatile("s_wait_asynccnt 0x0" ::: "memory");
        __syncthreads();

        // prefetch next tile into the alternate buffer (overlapped w/ compute)
        if (tile < 7) {
            const int nxt = cur ^ 1;
            const char* gs = (const char*)(sbase + (size_t)(tile + 1) * 16 * FCH);
            const char* gv = (const char*)(vbase + (size_t)(tile + 1) * 16 * FCH * 3);
            async_b128(ldsS[nxt] + tid * 16, gs + tid * 16);
            #pragma unroll
            for (int q = 0; q < 3; ++q)
                async_b128(ldsV[nxt] + (q * 256 + tid) * 16,
                           gv + (q * 256 + tid) * 16);
        }
        const float* curS = (const float*)(smem + (cur ? SM_S1 : SM_S0));
        const float* curV = (const float*)(smem + (cur ? SM_V1 : SM_V0));

        // --- my edge: sender = tile*16 + lmod (both halves duplicate) ---
        const int sme = tile * 16 + lmod;
        float dx = rx - sPos[sme * 3 + 0];
        float dy = ry - sPos[sme * 3 + 1];
        float dz = rz - sPos[sme * 3 + 2];
        float d2 = dx * dx + dy * dy + dz * dz + 1e-12f;
        float inv = fast_rsq(d2);            // 1/d  (v_rsq_f32)
        float d   = d2 * inv;                // d
        if (half == 0) {
            sUnitW[lmod * 4 + 0] = dx * inv;
            sUnitW[lmod * 4 + 1] = dy * inv;
            sUnitW[lmod * 4 + 2] = dz * inv;
        }

        // --- A fragment: phi (16 edges x K, K=8 live, rest zero) ---
        v16h aphi;
        #pragma unroll
        for (int p = 0; p < 8; ++p) {
            #pragma unroll
            for (int t = 0; t < 2; ++t) {
                int k = kslot(p, t, half);
                float ph = 0.0f;
                if (k < N_RBF) {
                    float dd = d - (float)k * cstep;
                    ph = fast_exp2(-dd * dd * g2);      // v_exp_f32
                }
                aphi[2 * p + t] = (_Float16)ph;
            }
        }

        // --- h = silu(phi @ W1 + b1), store to wave LDS in [m][f] ---
        #pragma unroll
        for (int nt = 0; nt < 4; ++nt) {
            v8f c = {};
            c = __builtin_amdgcn_wmma_f32_16x16x32_f16(
                    false, aphi, false, fw1[nt], (short)0, c, false, false);
            const int fcol = nt * 16 + lmod;
            const float bias = sB1[fcol];
            #pragma unroll
            for (int r = 0; r < 8; ++r) {
                float x  = c[r] + bias;
                float sg = fast_rcp(1.0f + fast_exp2(-LOG2E * x));
                float hs = x * sg;                       // silu
                int m = r + 8 * half;
                sHW[m * FCH + fcol] = (_Float16)hs;
            }
        }

        // --- re-layout h into A fragments (K=0..31, K=32..63);
        //     zero the self-edge row so all its rw outputs vanish ---
        v16h a0, a1;
        const bool self = (tile * 16 + lmod == j);
        #pragma unroll
        for (int p = 0; p < 8; ++p) {
            #pragma unroll
            for (int t = 0; t < 2; ++t) {
                int k = kslot(p, t, half);
                a0[2 * p + t] = self ? (_Float16)0.0f : sHW[lmod * FCH + k];
                a1[2 * p + t] = self ? (_Float16)0.0f : sHW[lmod * FCH + 32 + k];
            }
        }

        // --- unit vectors for my 8 D-rows (per-wave LDS, immediate offsets) ---
        float ux[8], uy[8], uz[8];
        #pragma unroll
        for (int r = 0; r < 8; ++r) {
            int m = r + 8 * half;
            ux[r] = sUnitW[m * 4 + 0];
            uy[r] = sUnitW[m * 4 + 1];
            uz[r] = sUnitW[m * 4 + 2];
        }

        // --- rw = h @ W2, ft-major: 3 independent WMMA chains per ft ---
        #pragma unroll
        for (int ft = 0; ft < 4; ++ft) {
            v8f c1 = {}, c2 = {}, c3 = {};
            c1 = __builtin_amdgcn_wmma_f32_16x16x32_f16(
                    false, a0, false, fw2[ft][0], (short)0, c1, false, false);
            c2 = __builtin_amdgcn_wmma_f32_16x16x32_f16(
                    false, a0, false, fw2[ft + 4][0], (short)0, c2, false, false);
            c3 = __builtin_amdgcn_wmma_f32_16x16x32_f16(
                    false, a0, false, fw2[ft + 8][0], (short)0, c3, false, false);
            c1 = __builtin_amdgcn_wmma_f32_16x16x32_f16(
                    false, a1, false, fw2[ft][1], (short)0, c1, false, false);
            c2 = __builtin_amdgcn_wmma_f32_16x16x32_f16(
                    false, a1, false, fw2[ft + 4][1], (short)0, c2, false, false);
            c3 = __builtin_amdgcn_wmma_f32_16x16x32_f16(
                    false, a1, false, fw2[ft + 8][1], (short)0, c3, false, false);

            const int fcol = ft * 16 + lmod;
            float as = 0.f, av0 = 0.f, av1 = 0.f, av2 = 0.f;
            #pragma unroll
            for (int r = 0; r < 8; ++r) {
                int mloc = r + 8 * half;                    // row in staged tile
                float sv = curS[mloc * FCH + fcol];         // shared r1/r3
                const float* vp = &curV[(mloc * FCH + fcol) * 3];
                float w3m = c3[r] * sv;                     // r3 * s
                as  += c1[r] * sv;                          // r1 * s
                av0 += c2[r] * vp[0] + w3m * ux[r];
                av1 += c2[r] * vp[1] + w3m * uy[r];
                av2 += c2[r] * vp[2] + w3m * uz[r];
            }
            acc_s[ft]    += as;
            acc_v[0][ft] += av0;
            acc_v[1][ft] += av1;
            acc_v[2][ft] += av2;
        }
    } // tile

    // --- cross-half reduce (rows 0-7 in lanes 0-15, rows 8-15 in 16-31) ---
    #pragma unroll
    for (int ft = 0; ft < 4; ++ft) {
        float ts = acc_s[ft]    + __shfl_xor(acc_s[ft],    16, 32);
        float t0 = acc_v[0][ft] + __shfl_xor(acc_v[0][ft], 16, 32);
        float t1 = acc_v[1][ft] + __shfl_xor(acc_v[1][ft], 16, 32);
        float t2 = acc_v[2][ft] + __shfl_xor(acc_v[2][ft], 16, 32);
        if (half == 0) {
            size_t sb = ((size_t)b * N_NODES + j) * FCH + ft * 16 + lmod;
            agg_s[sb] = ts * SC;
            agg_v[sb * 3 + 0] = t0 * SC;
            agg_v[sb * 3 + 1] = t1 * SC;
            agg_v[sb * 3 + 2] = t2 * SC;
        }
    }
}

// ---------------------------------------------------------------------------
// Kernel 3: node update  s' = s + agg_s @ ws ;  v' = v + einsum(agg_v, wv)
// ---------------------------------------------------------------------------
__global__ void macegnn_update(const float* __restrict__ s_in,
                               const float* __restrict__ v_in,
                               const float* __restrict__ agg_s,
                               const float* __restrict__ agg_v,
                               const float* __restrict__ wsM,   // [64,64]
                               const float* __restrict__ wvM,   // [64,64]
                               float* __restrict__ s_out,
                               float* __restrict__ v_out) {
    int idx = blockIdx.x * blockDim.x + threadIdx.x;
    if (idx >= B_SZ * N_NODES * FCH) return;
    int g = idx % FCH;
    size_t base = (size_t)(idx / FCH) * FCH;     // (b*N + n)*F
    float a = 0.f, av0 = 0.f, av1 = 0.f, av2 = 0.f;
    #pragma unroll 4
    for (int f = 0; f < FCH; ++f) {
        a += agg_s[base + f] * wsM[f * FCH + g];
        float w = wvM[f * FCH + g];
        const float* ap = agg_v + (base + f) * 3;
        av0 += ap[0] * w;
        av1 += ap[1] * w;
        av2 += ap[2] * w;
    }
    s_out[base + g] = s_in[base + g] + a;
    size_t vb = (base + g) * 3;
    v_out[vb + 0] = v_in[vb + 0] + av0;
    v_out[vb + 1] = v_in[vb + 1] + av1;
    v_out[vb + 2] = v_in[vb + 2] + av2;
}

// ---------------------------------------------------------------------------
// Kernel 4: equivariant readout  out = (einsum(v, wout) - pos) * scale
// ---------------------------------------------------------------------------
__global__ void macegnn_readout(const float* __restrict__ v,
                                const float* __restrict__ pos,
                                const float* __restrict__ wout,
                                const float* __restrict__ fscale,
                                float* __restrict__ out) {
    int idx = blockIdx.x * blockDim.x + threadIdx.x;
    if (idx >= B_SZ * N_NODES) return;
    size_t base = (size_t)idx * FCH;
    float o0 = 0.f, o1 = 0.f, o2 = 0.f;
    #pragma unroll 4
    for (int f = 0; f < FCH; ++f) {
        float w = wout[f];
        const float* vp = v + (base + f) * 3;
        o0 += vp[0] * w;
        o1 += vp[1] * w;
        o2 += vp[2] * w;
    }
    float sc = fscale[0];
    out[idx * 3 + 0] = (o0 - pos[idx * 3 + 0]) * sc;
    out[idx * 3 + 1] = (o1 - pos[idx * 3 + 1]) * sc;
    out[idx * 3 + 2] = (o2 - pos[idx * 3 + 2]) * sc;
}

// ---------------------------------------------------------------------------
extern "C" void kernel_launch(void* const* d_in, const int* in_sizes, int n_in,
                              void* d_out, int out_size, void* d_ws, size_t ws_size,
                              hipStream_t stream) {
    const float* pos     = (const float*)d_in[0];
    const int*   species = (const int*)d_in[1];
    const float* glob    = (const float*)d_in[2];
    const float* embed   = (const float*)d_in[3];
    const float* glob_w  = (const float*)d_in[4];
    const float* rbf_w1  = (const float*)d_in[5];   // [2,8,64]
    const float* rbf_b1  = (const float*)d_in[6];   // [2,64]
    const float* rbf_w2  = (const float*)d_in[7];   // [2,64,192]
    const float* wsW     = (const float*)d_in[8];   // [2,64,64]
    const float* wvW     = (const float*)d_in[9];   // [2,64,64]
    const float* wout    = (const float*)d_in[10];  // [64]
    const float* fscale  = (const float*)d_in[11];  // [1]

    float* wsf = (float*)d_ws;
    const size_t SN = (size_t)B_SZ * N_NODES * FCH;       // 262144
    float* s_a  = wsf;
    float* s_b  = wsf + SN;
    float* v_a  = wsf + 2 * SN;
    float* v_b  = wsf + 2 * SN + 3 * SN;
    float* aggs = wsf + 2 * SN + 6 * SN;
    float* aggv = wsf + 2 * SN + 6 * SN + SN;             // total 12*SN floats = 12 MiB

    macegnn_init<<<(int)(SN / 256), 256, 0, stream>>>(species, glob, embed, glob_w,
                                                      s_a, v_a);

    float* scur = s_a; float* snext = s_b;
    float* vcur = v_a; float* vnext = v_b;
    for (int l = 0; l < 2; ++l) {
        dim3 g(B_SZ, 16);
        macegnn_interact<<<g, 256, 0, stream>>>(
            pos, scur, vcur,
            rbf_w1 + (size_t)l * N_RBF * FCH,
            rbf_b1 + (size_t)l * FCH,
            rbf_w2 + (size_t)l * FCH * THREE_F,
            aggs, aggv);
        macegnn_update<<<(int)(SN / 256), 256, 0, stream>>>(
            scur, vcur, aggs, aggv,
            wsW + (size_t)l * FCH * FCH,
            wvW + (size_t)l * FCH * FCH,
            snext, vnext);
        float* t;
        t = scur; scur = snext; snext = t;
        t = vcur; vcur = vnext; vnext = t;
    }

    macegnn_readout<<<(B_SZ * N_NODES) / 256, 256, 0, stream>>>(
        vcur, pos, wout, fscale, (float*)d_out);
}
